// MultiHeadLinearAttention_3487513445062
// MI455X (gfx1250) — matrix operand order repaired and checked by reading, not compiled, not run on hardware
//
#include <hip/hip_runtime.h>
#include <hip/hip_bf16.h>

// ---------------------------------------------------------------------------
// MultiHeadLinearAttention for MI455X (gfx1250, wave32, WMMA).
// E=1024, H=16, D=64, B=4, NQ=NK=4096.  Heavy work = 4 GEMMs of
// [16384,1024]x[1024,1024] -> bf16 WMMA (v_wmma_f32_16x16x32_bf16),
// staged through LDS with async global->LDS copies when available.
// ---------------------------------------------------------------------------

#define E_DIM 1024
#define H_DIM 16
#define D_DIM 64
#define B_DIM 4
#define NQ_DIM 4096
#define NK_DIM 4096
#define M_ROWS (B_DIM * NQ_DIM)   // 16384

typedef __attribute__((ext_vector_type(16))) __bf16 v16bf;
typedef __attribute__((ext_vector_type(8)))  __bf16 v8bf;
typedef __attribute__((ext_vector_type(8)))  float  v8f;
typedef __attribute__((ext_vector_type(4)))  int    v4i;

#if defined(__gfx1250__) && \
    __has_builtin(__builtin_amdgcn_global_load_async_to_lds_b128) && \
    __has_builtin(__builtin_amdgcn_s_wait_asynccnt)
#define HAVE_ASYNC_LDS 1
#else
#define HAVE_ASYNC_LDS 0
#endif

#if HAVE_ASYNC_LDS
// global -> LDS async copy of 16 bytes (tracked by ASYNCcnt).
// Builtin params are int4* in global (AS1) / LDS (AS3) address spaces.
typedef __attribute__((address_space(1))) v4i gas_v4i;
typedef __attribute__((address_space(3))) v4i las_v4i;
__device__ inline void async_cp16(const __bf16* g, __bf16* l) {
    __builtin_amdgcn_global_load_async_to_lds_b128(
        (gas_v4i*)(g), (las_v4i*)(l), /*imm offset=*/0, /*cpol=*/0);
}
#endif

__device__ inline v16bf make_frag(const __bf16* p0, const __bf16* p1) {
    v8bf lo = *(const v8bf*)p0;
    v8bf hi = *(const v8bf*)p1;
    v16bf r;
#pragma unroll
    for (int i = 0; i < 8; ++i) { r[i] = lo[i]; r[i + 8] = hi[i]; }
    return r;
}

__device__ inline v8f wmma_bf16(v16bf a, v16bf b, v8f c) {
    return __builtin_amdgcn_wmma_f32_16x16x32_bf16(
        /*neg_a=*/false, a, /*neg_b=*/false, b,
        /*c_mod=*/(short)0, c, /*reuse_a=*/false, /*reuse_b=*/false);
}

// ---------------------------------------------------------------------------
// fp32 -> bf16 convert (grid-stride)
// ---------------------------------------------------------------------------
__global__ void cvt_f32_bf16_kernel(const float* __restrict__ src,
                                    __bf16* __restrict__ dst, long n) {
    long i = (long)blockIdx.x * blockDim.x + threadIdx.x;
    long stride = (long)gridDim.x * blockDim.x;
    for (; i < n; i += stride) dst[i] = (__bf16)src[i];
}

// ---------------------------------------------------------------------------
// GEMM: Y[M,N] = X[M,K] * W[N,K]^T + bias[N]   (bf16 in, f32 out)
// Block = 256 threads (8 waves). Block tile 128(M) x 128(N).
// Wave (wm 0..3, wn 0..1) computes 32(M) x 64(N) = 2x4 WMMA tiles.
// Double-buffered LDS; next K-tile streams in (async if available) while
// WMMAs consume the current one.  One barrier per K-step.
// ---------------------------------------------------------------------------
__global__ __launch_bounds__(256)
void gemm_bf16_wmma_kernel(const __bf16* __restrict__ X,
                           const __bf16* __restrict__ W,
                           const float* __restrict__ bias,
                           float* __restrict__ Y,
                           int M, int N, int K) {
    __shared__ alignas(16) __bf16 lds_a[2][128 * 32];
    __shared__ alignas(16) __bf16 lds_b[2][128 * 32];

    const int n0 = blockIdx.x * 128;
    const int m0 = blockIdx.y * 128;

    const int t    = threadIdx.x;
    const int wave = t >> 5;
    const int lane = t & 31;
    const int wm = wave >> 1;      // 0..3 : 32-row slab
    const int wn = wave & 1;       // 0..1 : 64-col slab
    const int lrow  = lane & 15;
    const int lhalf = lane >> 4;

    // staging coordinates: each thread copies 2x16B of A and 2x16B of B
    const int sr = t >> 1;               // 0..127 (row of A tile / row of W tile)
    const int sc = (t & 1) * 16;         // 0 or 16

    v8f z = {0.f, 0.f, 0.f, 0.f, 0.f, 0.f, 0.f, 0.f};
    v8f acc[2][4] = {{z, z, z, z}, {z, z, z, z}};

    auto stage = [&](int buf, int kb) {
        const __bf16* gA = X + (size_t)(m0 + sr) * K + kb + sc;
        const __bf16* gB = W + (size_t)(n0 + sr) * K + kb + sc;
        __bf16* lA = &lds_a[buf][sr * 32 + sc];
        __bf16* lB = &lds_b[buf][sr * 32 + sc];
#if HAVE_ASYNC_LDS
        async_cp16(gA,     lA);
        async_cp16(gA + 8, lA + 8);
        async_cp16(gB,     lB);
        async_cp16(gB + 8, lB + 8);
#else
        *(v8bf*)lA       = *(const v8bf*)gA;
        *(v8bf*)(lA + 8) = *(const v8bf*)(gA + 8);
        *(v8bf*)lB       = *(const v8bf*)gB;
        *(v8bf*)(lB + 8) = *(const v8bf*)(gB + 8);
        __builtin_prefetch(gA + 32, 0, 1);     // global_prefetch_b8
        __builtin_prefetch(gB + 32, 0, 1);
#endif
    };

    stage(0, 0);
    int buf = 0;
    for (int kb = 0; kb < K; kb += 32) {
#if HAVE_ASYNC_LDS
        __builtin_amdgcn_s_wait_asynccnt(0);
#endif
        __syncthreads();
        if (kb + 32 < K) stage(buf ^ 1, kb + 32);

        v16bf afrag[2], bfrag[4];
#pragma unroll
        for (int tm = 0; tm < 2; ++tm) {
            const int r = wm * 32 + tm * 16 + lrow;
            afrag[tm] = make_frag(&lds_a[buf][r * 32 + 8 * lhalf],
                                  &lds_a[buf][r * 32 + 16 + 8 * lhalf]);
        }
#pragma unroll
        for (int tn = 0; tn < 4; ++tn) {
            const int c = wn * 64 + tn * 16 + lrow;
            bfrag[tn] = make_frag(&lds_b[buf][c * 32 + 8 * lhalf],
                                  &lds_b[buf][c * 32 + 16 + 8 * lhalf]);
        }
#pragma unroll
        for (int tm = 0; tm < 2; ++tm)
#pragma unroll
            for (int tn = 0; tn < 4; ++tn)
                acc[tm][tn] = wmma_bf16(afrag[tm], bfrag[tn], acc[tm][tn]);
        buf ^= 1;
    }

    // C layout: lane col = lane&15, element r -> row r + 8*(lane>>4)
#pragma unroll
    for (int tm = 0; tm < 2; ++tm)
#pragma unroll
        for (int tn = 0; tn < 4; ++tn) {
            const int gn = n0 + wn * 64 + tn * 16 + lrow;
            const float bv = bias[gn];
#pragma unroll
            for (int r = 0; r < 8; ++r) {
                const int gm = m0 + wm * 32 + tm * 16 + r + 8 * lhalf;
                Y[(size_t)gm * N + gn] = acc[tm][tn][r] + bv;
            }
        }
}

// ---------------------------------------------------------------------------
// Fused LayerNorm + (elu(x)+1) -> bf16.  One block per row of 1024.
// ---------------------------------------------------------------------------
__global__ __launch_bounds__(256)
void ln_elu_kernel(const float* __restrict__ Y,
                   const float* __restrict__ g,
                   const float* __restrict__ beta,
                   __bf16* __restrict__ out) {
    __shared__ float red[256];
    const int row = blockIdx.x;
    const float* x = Y + (size_t)row * E_DIM;
    float s = 0.f, s2 = 0.f;
    float v[4];
#pragma unroll
    for (int j = 0; j < 4; ++j) {
        v[j] = x[threadIdx.x + 256 * j];
        s += v[j];
        s2 += v[j] * v[j];
    }
    red[threadIdx.x] = s;
    __syncthreads();
    for (int off = 128; off > 0; off >>= 1) {
        if (threadIdx.x < off) red[threadIdx.x] += red[threadIdx.x + off];
        __syncthreads();
    }
    const float mu = red[0] * (1.f / E_DIM);
    __syncthreads();
    red[threadIdx.x] = s2;
    __syncthreads();
    for (int off = 128; off > 0; off >>= 1) {
        if (threadIdx.x < off) red[threadIdx.x] += red[threadIdx.x + off];
        __syncthreads();
    }
    const float var = red[0] * (1.f / E_DIM) - mu * mu;
    const float rstd = rsqrtf(var + 1e-5f);
#pragma unroll
    for (int j = 0; j < 4; ++j) {
        const int i = threadIdx.x + 256 * j;
        float val = (v[j] - mu) * rstd * g[i] + beta[i];
        // elu(x)+1 : x>0 ? x+1 : exp(x)
        float r = (val > 0.f) ? (val + 1.f) : __expf(val);
        out[(size_t)row * E_DIM + i] = (__bf16)r;
    }
}

// ---------------------------------------------------------------------------
// kv_sum per (b,h):  kvT[e][d] = sum_n k[b,n,h,d] * v[b,n,h,e]  (bf16 out,
// stored transposed so the `num` GEMM's B-fragment reads are contiguous).
// One block (256 thr, 8 waves) per (b,h); 16 WMMA tiles, 2 per wave.
// ---------------------------------------------------------------------------
__global__ __launch_bounds__(256)
void kv_kernel(const __bf16* __restrict__ kbf,
               const __bf16* __restrict__ vbf,
               __bf16* __restrict__ kvT) {
    __shared__ alignas(16) __bf16 kT[64 * 32];   // [d][n]
    __shared__ alignas(16) __bf16 vT[64 * 32];   // [e][n]
    const int bh = blockIdx.x;
    const int b = bh / H_DIM, h = bh % H_DIM;
    const int t = threadIdx.x;
    const int wave = t >> 5, lane = t & 31;
    const int lrow = lane & 15, lhalf = lane >> 4;

    v8f z = {0.f, 0.f, 0.f, 0.f, 0.f, 0.f, 0.f, 0.f};
    v8f acc[2] = {z, z};

    const int sn = t >> 3;            // 0..31 : n within tile
    const int sd = (t & 7) * 8;       // 0,8,...,56 : d chunk

    for (int nb = 0; nb < NK_DIM; nb += 32) {
        const size_t base = ((size_t)b * NK_DIM + nb + sn) * E_DIM + h * D_DIM + sd;
        v8bf kc = *(const v8bf*)(kbf + base);
        v8bf vc = *(const v8bf*)(vbf + base);
#pragma unroll
        for (int j = 0; j < 8; ++j) {
            kT[(sd + j) * 32 + sn] = kc[j];
            vT[(sd + j) * 32 + sn] = vc[j];
        }
        __syncthreads();
#pragma unroll
        for (int i = 0; i < 2; ++i) {
            const int tile = wave * 2 + i;
            const int td = tile >> 2, te = tile & 3;
            const int rd = td * 16 + lrow;
            const int re = te * 16 + lrow;
            v16bf a = make_frag(&kT[rd * 32 + 8 * lhalf],
                                &kT[rd * 32 + 16 + 8 * lhalf]);
            v16bf bb = make_frag(&vT[re * 32 + 8 * lhalf],
                                 &vT[re * 32 + 16 + 8 * lhalf]);
            acc[i] = wmma_bf16(a, bb, acc[i]);
        }
        __syncthreads();
    }
#pragma unroll
    for (int i = 0; i < 2; ++i) {
        const int tile = wave * 2 + i;
        const int td = tile >> 2, te = tile & 3;
        const int e = te * 16 + lrow;
#pragma unroll
        for (int r = 0; r < 8; ++r) {
            const int d = td * 16 + r + 8 * lhalf;
            kvT[(size_t)bh * 4096 + (size_t)e * 64 + d] = (__bf16)acc[i][r];
        }
    }
}

// ---------------------------------------------------------------------------
// k_sum[b, h*64+d] = sum_n k[b,n,h*64+d]
// ---------------------------------------------------------------------------
__global__ void ksum_kernel(const __bf16* __restrict__ kbf,
                            float* __restrict__ ksum) {
    const int tid = blockIdx.x * blockDim.x + threadIdx.x;   // B*E threads
    if (tid >= B_DIM * E_DIM) return;
    const int b = tid / E_DIM, hd = tid % E_DIM;
    float s = 0.f;
    for (int n = 0; n < NK_DIM; ++n)
        s += (float)kbf[((size_t)b * NK_DIM + n) * E_DIM + hd];
    ksum[tid] = s;
}

// ---------------------------------------------------------------------------
// den[b,n,h] = q[b,n,h,:] . ksum[b,h,:] + 1e-8
// ---------------------------------------------------------------------------
__global__ void den_kernel(const __bf16* __restrict__ qbf,
                           const float* __restrict__ ksum,
                           float* __restrict__ den) {
    const int tid = blockIdx.x * blockDim.x + threadIdx.x;   // B*NQ*H
    if (tid >= B_DIM * NQ_DIM * H_DIM) return;
    const int h = tid % H_DIM;
    const int n = (tid / H_DIM) % NQ_DIM;
    const int b = tid / (H_DIM * NQ_DIM);
    const __bf16* q = qbf + ((size_t)b * NQ_DIM + n) * E_DIM + h * D_DIM;
    const float* ks = ksum + (size_t)b * E_DIM + h * D_DIM;
    float s = 0.f;
#pragma unroll 8
    for (int d = 0; d < D_DIM; ++d) s += (float)q[d] * ks[d];
    den[tid] = s + 1e-8f;
}

// ---------------------------------------------------------------------------
// num/den:  attn[b,n,h*64+e] = (sum_d q[b,n,h,d] * kvT[bh][e][d]) / den
// Grid: (NQ/128, B*H).  K-dim = 64 (2 WMMA steps), fragments straight from
// global (q rows and kvT rows are contiguous in the K index).
// ---------------------------------------------------------------------------
__global__ __launch_bounds__(256)
void num_kernel(const __bf16* __restrict__ qbf,
                const __bf16* __restrict__ kvT,
                const float* __restrict__ den,
                __bf16* __restrict__ attn) {
    const int bh = blockIdx.y;
    const int b = bh / H_DIM, h = bh % H_DIM;
    const int n0 = blockIdx.x * 128;
    const int t = threadIdx.x;
    const int wave = t >> 5, lane = t & 31;
    const int wm = wave >> 1, wn = wave & 1;
    const int lrow = lane & 15, lhalf = lane >> 4;

    v8f z = {0.f, 0.f, 0.f, 0.f, 0.f, 0.f, 0.f, 0.f};
    v8f acc[2][2] = {{z, z}, {z, z}};

#pragma unroll
    for (int kd = 0; kd < D_DIM; kd += 32) {
        v16bf afrag[2], bfrag[2];
#pragma unroll
        for (int tm = 0; tm < 2; ++tm) {
            const int row = n0 + wm * 32 + tm * 16 + lrow;
            const __bf16* p = qbf + ((size_t)b * NQ_DIM + row) * E_DIM
                            + h * D_DIM + kd;
            afrag[tm] = make_frag(p + 8 * lhalf, p + 16 + 8 * lhalf);
        }
#pragma unroll
        for (int tn = 0; tn < 2; ++tn) {
            const int e = wn * 32 + tn * 16 + lrow;
            const __bf16* p = kvT + (size_t)bh * 4096 + (size_t)e * 64 + kd;
            bfrag[tn] = make_frag(p + 8 * lhalf, p + 16 + 8 * lhalf);
        }
#pragma unroll
        for (int tm = 0; tm < 2; ++tm)
#pragma unroll
            for (int tn = 0; tn < 2; ++tn)
                acc[tm][tn] = wmma_bf16(afrag[tm], bfrag[tn], acc[tm][tn]);
    }

#pragma unroll
    for (int tm = 0; tm < 2; ++tm)
#pragma unroll
        for (int r = 0; r < 8; ++r) {
            const int row = n0 + wm * 32 + tm * 16 + r + 8 * lhalf;
            const float rcp = 1.f / den[((size_t)b * NQ_DIM + row) * H_DIM + h];
#pragma unroll
            for (int tn = 0; tn < 2; ++tn) {
                const int e = wn * 32 + tn * 16 + lrow;
                attn[((size_t)b * NQ_DIM + row) * E_DIM + h * D_DIM + e] =
                    (__bf16)(acc[tm][tn][r] * rcp);
            }
        }
}

// ---------------------------------------------------------------------------
// Host-side launch
// ---------------------------------------------------------------------------
static inline size_t align256(size_t x) { return (x + 255) & ~(size_t)255; }

extern "C" void kernel_launch(void* const* d_in, const int* in_sizes, int n_in,
                              void* d_out, int out_size, void* d_ws, size_t ws_size,
                              hipStream_t stream) {
    // inputs (fp32): 0 query, 1 key, 2 value, 3 Wq, 4 bq, 5 Wk, 6 bk,
    // 7 Wv, 8 bv, 9 Wo, 10 bo, 11 gq, 12 betaq, 13 gk, 14 betak
    const float* query = (const float*)d_in[0];
    const float* key   = (const float*)d_in[1];
    const float* value = (const float*)d_in[2];
    const float* Wq = (const float*)d_in[3];  const float* bq = (const float*)d_in[4];
    const float* Wk = (const float*)d_in[5];  const float* bk = (const float*)d_in[6];
    const float* Wv = (const float*)d_in[7];  const float* bv = (const float*)d_in[8];
    const float* Wo = (const float*)d_in[9];  const float* bo = (const float*)d_in[10];
    const float* gq = (const float*)d_in[11]; const float* betaq = (const float*)d_in[12];
    const float* gk = (const float*)d_in[13]; const float* betak = (const float*)d_in[14];
    float* out = (float*)d_out;

    const size_t ME = (size_t)M_ROWS * E_DIM;      // 16777216
    const size_t EE = (size_t)E_DIM * E_DIM;       // 1048576

    // workspace carve-out
    char* ws = (char*)d_ws;
    size_t off = 0;
    __bf16* qe_bf = (__bf16*)(ws + off); off = align256(off + ME * 2);   // -> q bf16 (reused)
    __bf16* ke_bf = (__bf16*)(ws + off); off = align256(off + ME * 2);   // -> k bf16 (reused)
    __bf16* ve_bf = (__bf16*)(ws + off); off = align256(off + ME * 2);   // -> v bf16 (reused)
    __bf16* Wq_bf = (__bf16*)(ws + off); off = align256(off + EE * 2);
    __bf16* Wk_bf = (__bf16*)(ws + off); off = align256(off + EE * 2);
    __bf16* Wv_bf = (__bf16*)(ws + off); off = align256(off + EE * 2);
    __bf16* Wo_bf = (__bf16*)(ws + off); off = align256(off + EE * 2);
    float*  proj  = (float*)(ws + off);  off = align256(off + ME * 4);   // GEMM scratch
    __bf16* kvT   = (__bf16*)(ws + off); off = align256(off + (size_t)B_DIM * H_DIM * 4096 * 2);
    float*  ksum  = (float*)(ws + off);  off = align256(off + (size_t)B_DIM * E_DIM * 4);
    float*  den   = (float*)(ws + off);  off = align256(off + (size_t)B_DIM * NQ_DIM * H_DIM * 4);
    __bf16* attn  = (__bf16*)(ws + off); off = align256(off + ME * 2);
    (void)ws_size; (void)in_sizes; (void)n_in; (void)out_size;

    // 1) fp32 -> bf16 converts
    cvt_f32_bf16_kernel<<<4096, 256, 0, stream>>>(query, qe_bf, (long)ME);
    cvt_f32_bf16_kernel<<<4096, 256, 0, stream>>>(key,   ke_bf, (long)ME);
    cvt_f32_bf16_kernel<<<4096, 256, 0, stream>>>(value, ve_bf, (long)ME);
    cvt_f32_bf16_kernel<<<1024, 256, 0, stream>>>(Wq, Wq_bf, (long)EE);
    cvt_f32_bf16_kernel<<<1024, 256, 0, stream>>>(Wk, Wk_bf, (long)EE);
    cvt_f32_bf16_kernel<<<1024, 256, 0, stream>>>(Wv, Wv_bf, (long)EE);
    cvt_f32_bf16_kernel<<<1024, 256, 0, stream>>>(Wo, Wo_bf, (long)EE);

    dim3 ggemm(E_DIM / 128, M_ROWS / 128);

    // 2) Q projection + LN + elu+1  (reuse qe_bf as q bf16)
    gemm_bf16_wmma_kernel<<<ggemm, 256, 0, stream>>>(qe_bf, Wq_bf, bq, proj,
                                                     M_ROWS, E_DIM, E_DIM);
    ln_elu_kernel<<<M_ROWS, 256, 0, stream>>>(proj, gq, betaq, qe_bf);

    // 3) K projection + LN + elu+1
    gemm_bf16_wmma_kernel<<<ggemm, 256, 0, stream>>>(ke_bf, Wk_bf, bk, proj,
                                                     M_ROWS, E_DIM, E_DIM);
    ln_elu_kernel<<<M_ROWS, 256, 0, stream>>>(proj, gk, betak, ke_bf);

    // 4) V projection (no LN), convert to bf16 in place of ve_bf
    gemm_bf16_wmma_kernel<<<ggemm, 256, 0, stream>>>(ve_bf, Wv_bf, bv, proj,
                                                     M_ROWS, E_DIM, E_DIM);
    cvt_f32_bf16_kernel<<<4096, 256, 0, stream>>>(proj, ve_bf, (long)ME);

    // 5) kv_sum (per b,h) and k_sum
    kv_kernel<<<B_DIM * H_DIM, 256, 0, stream>>>(ke_bf, ve_bf, kvT);
    ksum_kernel<<<(B_DIM * E_DIM + 255) / 256, 256, 0, stream>>>(ke_bf, ksum);

    // 6) denominator
    den_kernel<<<(B_DIM * NQ_DIM * H_DIM + 255) / 256, 256, 0, stream>>>(
        qe_bf, ksum, den);

    // 7) numerator / den -> attn (bf16)
    dim3 gnum(NQ_DIM / 128, B_DIM * H_DIM);
    num_kernel<<<gnum, 256, 0, stream>>>(qe_bf, kvT, den, attn);

    // 8) output projection -> d_out (f32)
    gemm_bf16_wmma_kernel<<<ggemm, 256, 0, stream>>>(attn, Wo_bf, bo, out,
                                                     M_ROWS, E_DIM, E_DIM);
}
